// GraphConvModel_10342281249157
// MI455X (gfx1250) — matrix operand order repaired and checked by reading, compiled, not run on hardware
//
#include <hip/hip_runtime.h>

// ---------------- problem constants (match reference) ----------------
#define N_NODES  50000
#define N_EDGES  800000
#define F_IN     64
#define HID      512
#define N_GRAPHS 32
#define N_CLASSES 24

typedef __bf16 bf16;
typedef __attribute__((ext_vector_type(16))) __bf16 v16bf;
typedef __attribute__((ext_vector_type(8)))  float  v8f;

union Frag16 {            // one WMMA 16-bf16 operand fragment per lane (8 VGPRs)
    v16bf  v;
    float4 f[2];          // two 16-byte loads
};

// ---------------- utility kernels ----------------
__global__ void zero_f32(float* __restrict__ p, int n) {
    int i = blockIdx.x * blockDim.x + threadIdx.x;
    if (i < n) p[i] = 0.0f;
}

__global__ void f32_to_bf16(const float* __restrict__ in, bf16* __restrict__ out, int n) {
    int i = blockIdx.x * blockDim.x + threadIdx.x;
    if (i < n) out[i] = (bf16)in[i];
}

// w[K][Ncols] f32 (row-major, fan_in x fan_out)  ->  wt[Ncols][K] bf16
__global__ void transpose_to_bf16(const float* __restrict__ w, bf16* __restrict__ wt,
                                  int K, int Ncols) {
    int i = blockIdx.x * blockDim.x + threadIdx.x;
    if (i >= K * Ncols) return;
    int n = i / K, k = i % K;
    wt[i] = (bf16)w[k * Ncols + n];
}

// ---------------- message passing ----------------
__global__ void compute_deg(const int* __restrict__ dst, float* __restrict__ deg) {
    int e = blockIdx.x * blockDim.x + threadIdx.x;
    if (e < N_EDGES) atomicAdd(&deg[dst[e]], 1.0f);
}

// layer-1 scatter: x is f32 [N, 64]; 4 floats per thread (float4 gather)
__global__ void scatter_x_f32(const float* __restrict__ x,
                              const int* __restrict__ src, const int* __restrict__ dst,
                              float* __restrict__ agg) {
    int idx = blockIdx.x * blockDim.x + threadIdx.x;     // N_EDGES * 16 threads
    if (idx >= N_EDGES * 16) return;
    int e  = idx >> 4;
    int f4 = (idx & 15) * 4;
    int s = src[e], d = dst[e];
    float4 v = *(const float4*)(x + (size_t)s * F_IN + f4);
    float* o = agg + (size_t)d * F_IN + f4;
    atomicAdd(o + 0, v.x); atomicAdd(o + 1, v.y);
    atomicAdd(o + 2, v.z); atomicAdd(o + 3, v.w);
}

// layers 2/3 scatter: h is bf16 [N, 512] (51 MB -> L2-resident); 8 bf16 per thread
__global__ void scatter_h_bf16(const bf16* __restrict__ h,
                               const int* __restrict__ src, const int* __restrict__ dst,
                               float* __restrict__ agg) {
    int idx = blockIdx.x * blockDim.x + threadIdx.x;     // N_EDGES * 64 threads
    if (idx >= N_EDGES * 64) return;
    int e  = idx >> 6;
    int f8 = (idx & 63) * 8;
    int s = src[e], d = dst[e];
    float4 raw = *(const float4*)(h + (size_t)s * HID + f8);   // 8 bf16 = 16 B
    const bf16* bp = (const bf16*)&raw;
    float* o = agg + (size_t)d * HID + f8;
#pragma unroll
    for (int i = 0; i < 8; ++i) atomicAdd(o + i, (float)bp[i]);
}

// agg f32 -> bf16 (optionally divide by degree for 'mean' aggregation)
__global__ void finalize_agg(const float* __restrict__ agg, const float* __restrict__ deg,
                             bf16* __restrict__ out, int total, int K, int use_mean) {
    int i = blockIdx.x * blockDim.x + threadIdx.x;
    if (i >= total) return;
    float v = agg[i];
    if (use_mean) v /= fmaxf(deg[i / K], 1.0f);
    out[i] = (bf16)v;
}

// ---------------- WMMA GEMM:  out = relu(A0 @ B0^T + A1 @ B1^T + bias) ----------------
// A0 = aggregated messages [N, K] bf16, A1 = node features [N, K] bf16,
// B0 = w_rel^T [HID, K] bf16, B1 = w_root^T [HID, K] bf16, out = [N, HID] bf16.
// Block = 256 threads = 8 waves; block owns 16 rows, each wave a 64-col slab
// (4 accumulator tiles).  50000 = 3125*16 exactly -> no bounds checks, EXEC all-ones.
__global__ void __launch_bounds__(256)
gemm_wmma_bf16(const bf16* __restrict__ A0, const bf16* __restrict__ A1,
               const bf16* __restrict__ B0, const bf16* __restrict__ B1,
               const float* __restrict__ bias, bf16* __restrict__ out, int K) {
    const int lane = threadIdx.x & 31;
    const int wave = threadIdx.x >> 5;
    const int m0   = blockIdx.x * 16;
    const int c0   = wave * 64;
    const int mrow = lane & 15;
    const int half = lane >> 4;

    v8f acc[4] = {};

    const bf16* Ap[2] = { A0, A1 };
    const bf16* Bp[2] = { B0, B1 };

#pragma unroll
    for (int p = 0; p < 2; ++p) {
        const bf16* arow = Ap[p] + (size_t)(m0 + mrow) * K;
        const bf16* Bm   = Bp[p];
        for (int k0 = 0; k0 < K; k0 += 32) {
            // A fragment: ISA 16-bit A 16x32 layout (two 16B runs per lane)
            Frag16 a;
            a.f[0] = *(const float4*)(arow + k0 + half * 8);
            a.f[1] = *(const float4*)(arow + k0 + 16 + half * 8);
#pragma unroll
            for (int t = 0; t < 4; ++t) {
                // B fragment: lane holds K = half*8 .. +7 and 16+half*8 .. +7 of column (c0+t*16+n)
                const bf16* brow = Bm + (size_t)(c0 + t * 16 + mrow) * K + k0 + half * 8;
                Frag16 b;
                b.f[0] = *(const float4*)(brow);
                b.f[1] = *(const float4*)(brow + 16);
                acc[t] = __builtin_amdgcn_wmma_f32_16x16x32_bf16(
                    false, a.v, false, b.v, (short)0, acc[t], false, false);
            }
        }
    }

    // epilogue: bias + ReLU + bf16 store.  C/D layout: VGPR v -> row m0+v+half*8.
#pragma unroll
    for (int t = 0; t < 4; ++t) {
        int col = c0 + t * 16 + mrow;
        float bv = bias[col];
#pragma unroll
        for (int v = 0; v < 8; ++v) {
            int row = m0 + v + half * 8;
            float val = fmaxf(acc[t][v] + bv, 0.0f);
            out[(size_t)row * HID + col] = (bf16)val;
        }
    }
}

// ---------------- pooling + head ----------------
__global__ void pool_sum(const bf16* __restrict__ h, const int* __restrict__ batch,
                         float* __restrict__ pooled) {
    int i = blockIdx.x * blockDim.x + threadIdx.x;      // N_NODES * HID
    if (i >= N_NODES * HID) return;
    int n = i / HID, f = i % HID;
    atomicAdd(&pooled[batch[n] * HID + f], (float)h[i]);
}

__global__ void count_nodes(const int* __restrict__ batch, float* __restrict__ counts) {
    int n = blockIdx.x * blockDim.x + threadIdx.x;
    if (n < N_NODES) atomicAdd(&counts[batch[n]], 1.0f);
}

__global__ void head_kernel(const float* __restrict__ pooled, const float* __restrict__ counts,
                            const float* __restrict__ w_out, const float* __restrict__ b_out,
                            float* __restrict__ out) {
    int tid = blockIdx.x * blockDim.x + threadIdx.x;    // 32*24 = 768
    if (tid >= N_GRAPHS * N_CLASSES) return;
    int g = tid / N_CLASSES, c = tid % N_CLASSES;
    float inv = 1.0f / fmaxf(counts[g], 1.0f);
    float s = 0.0f;
    for (int k = 0; k < HID; ++k) s += pooled[g * HID + k] * w_out[k * N_CLASSES + c];
    out[tid] = s * inv + b_out[c];
}

// ---------------- launch ----------------
static inline unsigned cdiv(long long a, long long b) { return (unsigned)((a + b - 1) / b); }

extern "C" void kernel_launch(void* const* d_in, const int* in_sizes, int n_in,
                              void* d_out, int out_size, void* d_ws, size_t ws_size,
                              hipStream_t stream) {
    (void)in_sizes; (void)n_in; (void)out_size; (void)ws_size;

    const float* x       = (const float*)d_in[0];
    const int*   ei      = (const int*)d_in[1];
    const int*   batch   = (const int*)d_in[2];
    const float* w1_rel  = (const float*)d_in[3];
    const float* b1      = (const float*)d_in[4];
    const float* w1_root = (const float*)d_in[5];
    const float* w2_rel  = (const float*)d_in[6];
    const float* b2      = (const float*)d_in[7];
    const float* w2_root = (const float*)d_in[8];
    const float* w3_rel  = (const float*)d_in[9];
    const float* b3      = (const float*)d_in[10];
    const float* w3_root = (const float*)d_in[11];
    const float* w_out   = (const float*)d_in[12];
    const float* b_out   = (const float*)d_in[13];
    const int* src = ei;
    const int* dst = ei + N_EDGES;

    // workspace carve (256B aligned)
    char* ws = (char*)d_ws;
    auto carve = [&](size_t bytes) { char* p = ws; ws += (bytes + 255) & ~(size_t)255; return p; };
    float* agg    = (float*)carve((size_t)N_NODES * HID * 4);   // f32 accumulation (102 MB, L2-resident)
    bf16*  hA     = (bf16*) carve((size_t)N_NODES * HID * 2);   // feature ping
    bf16*  hB     = (bf16*) carve((size_t)N_NODES * HID * 2);   // feature pong
    bf16*  aggB   = (bf16*) carve((size_t)N_NODES * HID * 2);   // bf16 aggregated msgs
    bf16*  xB     = (bf16*) carve((size_t)N_NODES * F_IN * 2);
    bf16*  wrelT  = (bf16*) carve((size_t)HID * HID * 2);
    bf16*  wrootT = (bf16*) carve((size_t)HID * HID * 2);
    float* deg    = (float*)carve((size_t)N_NODES * 4);
    float* pooled = (float*)carve((size_t)N_GRAPHS * HID * 4);
    float* counts = (float*)carve((size_t)N_GRAPHS * 4);

    const int B = 256;

    // shared prep: degrees (same graph for layers 2 & 3), x in bf16
    zero_f32<<<cdiv(N_NODES, B), B, 0, stream>>>(deg, N_NODES);
    compute_deg<<<cdiv(N_EDGES, B), B, 0, stream>>>(dst, deg);
    f32_to_bf16<<<cdiv((long long)N_NODES * F_IN, B), B, 0, stream>>>(x, xB, N_NODES * F_IN);

    // -------- layer 1 (aggr = add, K = 64) --------
    transpose_to_bf16<<<cdiv(F_IN * HID, B), B, 0, stream>>>(w1_rel,  wrelT,  F_IN, HID);
    transpose_to_bf16<<<cdiv(F_IN * HID, B), B, 0, stream>>>(w1_root, wrootT, F_IN, HID);
    zero_f32<<<cdiv((long long)N_NODES * F_IN, B), B, 0, stream>>>(agg, N_NODES * F_IN);
    scatter_x_f32<<<cdiv((long long)N_EDGES * 16, B), B, 0, stream>>>(x, src, dst, agg);
    finalize_agg<<<cdiv((long long)N_NODES * F_IN, B), B, 0, stream>>>(agg, deg, aggB,
                                                                       N_NODES * F_IN, F_IN, 0);
    gemm_wmma_bf16<<<N_NODES / 16, 256, 0, stream>>>(aggB, xB, wrelT, wrootT, b1, hA, F_IN);

    // -------- layer 2 (aggr = mean, K = 512) --------
    transpose_to_bf16<<<cdiv(HID * HID, B), B, 0, stream>>>(w2_rel,  wrelT,  HID, HID);
    transpose_to_bf16<<<cdiv(HID * HID, B), B, 0, stream>>>(w2_root, wrootT, HID, HID);
    zero_f32<<<cdiv((long long)N_NODES * HID, B), B, 0, stream>>>(agg, N_NODES * HID);
    scatter_h_bf16<<<cdiv((long long)N_EDGES * 64, B), B, 0, stream>>>(hA, src, dst, agg);
    finalize_agg<<<cdiv((long long)N_NODES * HID, B), B, 0, stream>>>(agg, deg, aggB,
                                                                      N_NODES * HID, HID, 1);
    gemm_wmma_bf16<<<N_NODES / 16, 256, 0, stream>>>(aggB, hA, wrelT, wrootT, b2, hB, HID);

    // -------- layer 3 (aggr = mean, K = 512) --------
    transpose_to_bf16<<<cdiv(HID * HID, B), B, 0, stream>>>(w3_rel,  wrelT,  HID, HID);
    transpose_to_bf16<<<cdiv(HID * HID, B), B, 0, stream>>>(w3_root, wrootT, HID, HID);
    zero_f32<<<cdiv((long long)N_NODES * HID, B), B, 0, stream>>>(agg, N_NODES * HID);
    scatter_h_bf16<<<cdiv((long long)N_EDGES * 64, B), B, 0, stream>>>(hB, src, dst, agg);
    finalize_agg<<<cdiv((long long)N_NODES * HID, B), B, 0, stream>>>(agg, deg, aggB,
                                                                      N_NODES * HID, HID, 1);
    gemm_wmma_bf16<<<N_NODES / 16, 256, 0, stream>>>(aggB, hB, wrelT, wrootT, b3, hA, HID);

    // -------- global mean pool + linear head --------
    zero_f32<<<cdiv(N_GRAPHS * HID, B), B, 0, stream>>>(pooled, N_GRAPHS * HID);
    zero_f32<<<1, B, 0, stream>>>(counts, N_GRAPHS);
    pool_sum<<<cdiv((long long)N_NODES * HID, B), B, 0, stream>>>(hA, batch, pooled);
    count_nodes<<<cdiv(N_NODES, B), B, 0, stream>>>(batch, counts);
    head_kernel<<<cdiv(N_GRAPHS * N_CLASSES, B), B, 0, stream>>>(pooled, counts, w_out, b_out,
                                                                 (float*)d_out);
}
